// Head_71313636982964
// MI455X (gfx1250) — matrix operand order repaired and checked by reading, not compiled, not run on hardware
//
#include <hip/hip_runtime.h>
#include <hip/hip_bf16.h>

#define NB 8
#define TT 2048
#define NE 1024
#define HS 128
#define TQ 64     // query rows per block
#define KC 64     // key chunk (kernel 2)

// workspace layout (bf16 elements)
#define K0  ((size_t)0)                    // k   [b][t][h]
#define Q0  ((size_t)(NB) * TT * HS)       // q   [b][t][h], pre-scaled
#define VT0 ((size_t)(2 * NB) * TT * HS)   // v^T [b][h][t]
#define WT0 ((size_t)(3 * NB) * TT * HS)   // W^T [which][col][k], bf16

typedef __bf16 bf16;
typedef __attribute__((ext_vector_type(16))) __bf16 v16bf;
typedef __attribute__((ext_vector_type(8)))  __bf16 v8bf;
typedef __attribute__((ext_vector_type(8)))  float  v8f;
typedef __attribute__((ext_vector_type(4)))  int    v4i;

#define AS1 __attribute__((address_space(1)))
#define AS3 __attribute__((address_space(3)))

#ifndef __has_builtin
#define __has_builtin(x) 0
#endif
#if __has_builtin(__builtin_amdgcn_global_load_async_to_lds_b128)
#define HAVE_ASYNC_LDS 1
#else
#define HAVE_ASYNC_LDS 0
#endif

__device__ __forceinline__ void wait_async() {
#if HAVE_ASYNC_LDS
#if __has_builtin(__builtin_amdgcn_s_wait_asynccnt)
  __builtin_amdgcn_s_wait_asynccnt(0);
#else
  asm volatile("s_wait_asynccnt 0x0" ::: "memory");
#endif
#endif
}

// wave-local LDS ordering fence (LDS ops are in-order per wave; this stops
// compiler reordering and drains DScnt)
__device__ __forceinline__ void lds_fence() {
  asm volatile("s_wait_dscnt 0x0" ::: "memory");
}

// A-matrix fragment (16x32 bf16), row-major source with leading dim `ld`.
// lanes 0-15: row=lane, K=0..15; lanes 16-31: same rows, K=16..31.
__device__ __forceinline__ v16bf load_a_frag(const bf16* __restrict__ src, int ld,
                                             int row_base, int kbase) {
  const int l = threadIdx.x & 31;
  const bf16* p = src + (size_t)(row_base + (l & 15)) * ld + kbase + ((l >> 4) << 4);
  return *(const v16bf*)(p);
}

// B-matrix fragment (32x16 bf16) from "N-major" storage src[n][k] (k contiguous).
// lane = column N; lanes 0-15 hold K=0..15, lanes 16-31 hold K=16..31.
__device__ __forceinline__ v16bf load_b_frag(const bf16* __restrict__ src, int ld,
                                             int n_base, int kbase) {
  const int l = threadIdx.x & 31;
  const bf16* p = src + (size_t)(n_base + (l & 15)) * ld + kbase + ((l >> 4) << 4);
  return *(const v16bf*)(p);
}

// ---------------- Kernel 0: W fp32 -> bf16, transposed to [col][k] ----------------
__global__ void __launch_bounds__(256) wconv_kernel(
    const float* __restrict__ Wk, const float* __restrict__ Wq,
    const float* __restrict__ Wv, bf16* __restrict__ ws) {
  const int which = blockIdx.y;
  const float* W = (which == 0) ? Wk : (which == 1) ? Wq : Wv;
  bf16* wt = ws + WT0 + (size_t)which * (HS * NE);
  int e = blockIdx.x * 256 + threadIdx.x;   // e < NE*HS, row-major index of W
  int col = e & (HS - 1);
  int k = e >> 7;
  wt[(size_t)col * NE + k] = (bf16)W[e];
}

// ---------------- Kernel 1: QKV projection (double-buffered, async W staging) ------
__device__ __forceinline__ void stage_x(bf16* buf, const float* __restrict__ x,
                                        int row0, int tid, int kb) {
  const int rr = tid >> 1;
  const int kh = (tid & 1) << 4;
  const float* gp = x + (size_t)(row0 + rr) * NE + kb + kh;
  v8bf t0, t1;
#pragma unroll
  for (int j = 0; j < 8; ++j) { t0[j] = (bf16)gp[j]; t1[j] = (bf16)gp[8 + j]; }
  *(v8bf*)(buf + rr * 32 + kh)     = t0;
  *(v8bf*)(buf + rr * 32 + kh + 8) = t1;
}

__device__ __forceinline__ void stage_w(bf16* buf, const bf16* __restrict__ wt,
                                        int tid, int kb) {
#if HAVE_ASYNC_LDS
#pragma unroll
  for (int i0 = 0; i0 < 32; i0 += 8)
    __builtin_amdgcn_global_load_async_to_lds_b128(
        (AS1 v4i*)(wt + (size_t)tid * NE + kb + i0),
        (AS3 v4i*)(buf + tid * 32 + i0), 0, 0);
#else
#pragma unroll
  for (int i0 = 0; i0 < 32; i0 += 8)
    *(v8bf*)(buf + tid * 32 + i0) = *(const v8bf*)(wt + (size_t)tid * NE + kb + i0);
#endif
}

__global__ void __launch_bounds__(128) qkv_kernel(
    const float* __restrict__ x, bf16* __restrict__ ws) {
  __shared__ __align__(16) bf16 sx[2][TQ * 32];   // x chunk   [row][k]
  __shared__ __align__(16) bf16 sw[2][HS * 32];   // W^T chunk [col][k]

  const int tid = threadIdx.x;
  const int wv  = tid >> 5;
  const int row0  = blockIdx.x * TQ;
  const int which = blockIdx.y;                   // 0=k, 1=q, 2=v
  const bf16* wt = ws + WT0 + (size_t)which * (HS * NE);

  v8f acc[8];
#pragma unroll
  for (int i = 0; i < 8; ++i)
#pragma unroll
    for (int j = 0; j < 8; ++j) acc[i][j] = 0.0f;

  stage_x(sx[0], x, row0, tid, 0);
  stage_w(sw[0], wt, tid, 0);

  for (int it = 0; it < NE / 32; ++it) {
    wait_async();
    __syncthreads();                       // chunk `it` visible to all waves
    if (it + 1 < NE / 32) {                // prefetch next chunk into other buffer
      stage_x(sx[(it + 1) & 1], x, row0, tid, (it + 1) * 32);
      stage_w(sw[(it + 1) & 1], wt, tid, (it + 1) * 32);
    }
    const bf16* cx = sx[it & 1];
    const bf16* cw = sw[it & 1];
    v16bf a = load_a_frag(cx, 32, 16 * wv, 0);
    v16bf bfr[8];
#pragma unroll
    for (int nt = 0; nt < 8; ++nt) bfr[nt] = load_b_frag(cw, 32, 16 * nt, 0);
#pragma unroll
    for (int nt = 0; nt < 8; ++nt)
      acc[nt] = __builtin_amdgcn_wmma_f32_16x16x32_bf16(false, a, false, bfr[nt],
                                                        (short)0, acc[nt], false, false);
  }

  const int l = tid & 31;
  const int cl = l & 15;
  const int rhalf = (l >> 4) << 3;
  if (which == 2) {
    // store V transposed: vT[b][h][t]
    const int bidx = row0 >> 11;                       // blocks never straddle batches
    const int key0 = (row0 & (TT - 1)) + 16 * wv + rhalf;
    bf16* vt = ws + VT0 + (size_t)bidx * HS * TT;
#pragma unroll
    for (int nt = 0; nt < 8; ++nt)
#pragma unroll
      for (int v = 0; v < 8; ++v)
        vt[(size_t)(16 * nt + cl) * TT + key0 + v] = (bf16)acc[nt][v];
  } else {
    const float scale = (which == 1) ? 0.08838834764831845f : 1.0f;  // 128^-0.5 into q
    bf16* outb = ws + (size_t)which * (NB * TT * HS);
#pragma unroll
    for (int nt = 0; nt < 8; ++nt)
#pragma unroll
      for (int v = 0; v < 8; ++v) {
        int grow = row0 + 16 * wv + rhalf + v;
        outb[(size_t)grow * HS + 16 * nt + cl] = (bf16)(acc[nt][v] * scale);
      }
  }
}

// ---------------- Kernel 2: fused causal flash attention (barrier-free) ----------
__global__ void __launch_bounds__(128) attn_kernel(
    const bf16* __restrict__ ws, float* __restrict__ out) {
  __shared__ __align__(16) bf16 pbuf[4][16 * KC];  // per-wave P relay (C->A layout)

  const int tid = threadIdx.x;
  const int wv  = tid >> 5;
  const int l   = tid & 31;
  const int q0  = blockIdx.x * TQ;
  const int b   = blockIdx.y;

  const bf16* kb_ = ws + K0  + (size_t)b * TT * HS;   // keys    [t][h]
  const bf16* qb_ = ws + Q0  + (size_t)b * TT * HS;   // queries [t][h] (pre-scaled)
  const bf16* vt_ = ws + VT0 + (size_t)b * HS * TT;   // values  [h][t]

  const int qr0 = q0 + 16 * wv;                       // this wave's 16 query rows
  const int cl = l & 15;
  const int rhalf = (l >> 4) << 3;

  v16bf qf[4];
#pragma unroll
  for (int s = 0; s < 4; ++s) qf[s] = load_a_frag(qb_, HS, qr0, 32 * s);

  v8f o[8];
  float m[8], lsum[8];
#pragma unroll
  for (int i = 0; i < 8; ++i) {
#pragma unroll
    for (int j = 0; j < 8; ++j) o[i][j] = 0.0f;
    m[i] = -__builtin_inff();
    lsum[i] = 0.0f;
  }

  // per-wave causal bound: keys 0 .. qr0+15
  for (int kc = 0; kc < qr0 + 16; kc += KC) {
    if (kc + KC < qr0 + 16) {  // hint next chunk toward the caches
      __builtin_prefetch(kb_ + (size_t)(kc + KC + l) * HS, 0, 0);
      __builtin_prefetch(vt_ + (size_t)l * TT + kc + KC, 0, 0);
    }

    // S = Q * K^T : per k-step, 4 independent column tiles
    v8f s[4];
#pragma unroll
    for (int ct = 0; ct < 4; ++ct)
#pragma unroll
      for (int j = 0; j < 8; ++j) s[ct][j] = 0.0f;
#pragma unroll
    for (int st = 0; st < 4; ++st) {
      v16bf k0 = load_b_frag(kb_, HS, kc +  0, 32 * st);
      v16bf k1 = load_b_frag(kb_, HS, kc + 16, 32 * st);
      v16bf k2 = load_b_frag(kb_, HS, kc + 32, 32 * st);
      v16bf k3 = load_b_frag(kb_, HS, kc + 48, 32 * st);
      s[0] = __builtin_amdgcn_wmma_f32_16x16x32_bf16(false, qf[st], false, k0, (short)0, s[0], false, false);
      s[1] = __builtin_amdgcn_wmma_f32_16x16x32_bf16(false, qf[st], false, k1, (short)0, s[1], false, false);
      s[2] = __builtin_amdgcn_wmma_f32_16x16x32_bf16(false, qf[st], false, k2, (short)0, s[2], false, false);
      s[3] = __builtin_amdgcn_wmma_f32_16x16x32_bf16(false, qf[st], false, k3, (short)0, s[3], false, false);
    }

    // causal mask only on chunks crossing the diagonal (wave-uniform branch)
    if (kc + KC - 1 > qr0) {
#pragma unroll
      for (int ct = 0; ct < 4; ++ct) {
        int col = kc + 16 * ct + cl;
#pragma unroll
        for (int v = 0; v < 8; ++v)
          if (col > qr0 + rhalf + v) s[ct][v] = -3.0e38f;
      }
    }

    // online softmax: row max via 16-lane half reductions
    float mnew[8], cf[8];
#pragma unroll
    for (int v = 0; v < 8; ++v) {
      float t = fmaxf(fmaxf(s[0][v], s[1][v]), fmaxf(s[2][v], s[3][v]));
#pragma unroll
      for (int msk = 1; msk <= 8; msk <<= 1)
        t = fmaxf(t, __shfl_xor(t, msk, 32));
      mnew[v] = fmaxf(m[v], t);
      cf[v] = __expf(m[v] - mnew[v]);
    }

    // P = exp(S - m) -> LDS relay; accumulate row sums
    float rs[8];
#pragma unroll
    for (int v = 0; v < 8; ++v) rs[v] = 0.0f;
    lds_fence();  // WAR: previous iteration's P loads complete before overwrite
#pragma unroll
    for (int ct = 0; ct < 4; ++ct)
#pragma unroll
      for (int v = 0; v < 8; ++v) {
        float p = __expf(s[ct][v] - mnew[v]);
        rs[v] += p;
        pbuf[wv][(rhalf + v) * KC + 16 * ct + cl] = (bf16)p;
      }
    lds_fence();  // RAW: P stores ordered before cross-lane fragment loads
#pragma unroll
    for (int v = 0; v < 8; ++v) {
      float t = rs[v];
#pragma unroll
      for (int msk = 1; msk <= 8; msk <<= 1)
        t += __shfl_xor(t, msk, 32);
      lsum[v] = lsum[v] * cf[v] + t;
      m[v] = mnew[v];
    }
#pragma unroll
    for (int nt = 0; nt < 8; ++nt)
#pragma unroll
      for (int v = 0; v < 8; ++v) o[nt][v] *= cf[v];

    // O += P * V  (V^T streamed from global, contiguous per lane)
#pragma unroll
    for (int st2 = 0; st2 < 2; ++st2) {
      v16bf pa = load_a_frag(&pbuf[wv][0], KC, 0, 32 * st2);
#pragma unroll
      for (int g = 0; g < 2; ++g) {
        v16bf f0 = load_b_frag(vt_, TT, 16 * (4 * g + 0), kc + 32 * st2);
        v16bf f1 = load_b_frag(vt_, TT, 16 * (4 * g + 1), kc + 32 * st2);
        v16bf f2 = load_b_frag(vt_, TT, 16 * (4 * g + 2), kc + 32 * st2);
        v16bf f3 = load_b_frag(vt_, TT, 16 * (4 * g + 3), kc + 32 * st2);
        o[4 * g + 0] = __builtin_amdgcn_wmma_f32_16x16x32_bf16(false, pa, false, f0, (short)0, o[4 * g + 0], false, false);
        o[4 * g + 1] = __builtin_amdgcn_wmma_f32_16x16x32_bf16(false, pa, false, f1, (short)0, o[4 * g + 1], false, false);
        o[4 * g + 2] = __builtin_amdgcn_wmma_f32_16x16x32_bf16(false, pa, false, f2, (short)0, o[4 * g + 2], false, false);
        o[4 * g + 3] = __builtin_amdgcn_wmma_f32_16x16x32_bf16(false, pa, false, f3, (short)0, o[4 * g + 3], false, false);
      }
    }
  }

  // epilogue: normalize by row sums, write fp32
#pragma unroll
  for (int v = 0; v < 8; ++v) {
    float inv = 1.0f / lsum[v];
    int grow = qr0 + rhalf + v;
#pragma unroll
    for (int nt = 0; nt < 8; ++nt)
      out[((size_t)b * TT + grow) * HS + 16 * nt + cl] = o[nt][v] * inv;
  }
}

extern "C" void kernel_launch(void* const* d_in, const int* in_sizes, int n_in,
                              void* d_out, int out_size, void* d_ws, size_t ws_size,
                              hipStream_t stream) {
  const float* x  = (const float*)d_in[0];
  const float* Wk = (const float*)d_in[1];
  const float* Wq = (const float*)d_in[2];
  const float* Wv = (const float*)d_in[3];
  bf16* ws   = (bf16*)d_ws;   // needs (3*NB*TT*HS + 3*HS*NE) * 2 bytes ~= 12.75 MB
  float* out = (float*)d_out;

  wconv_kernel<<<dim3((NE * HS) / 256, 3), 256, 0, stream>>>(Wk, Wq, Wv, ws);
  qkv_kernel<<<dim3((NB * TT) / TQ, 3), 128, 0, stream>>>(x, ws);
  attn_kernel<<<dim3(TT / TQ, NB), 128, 0, stream>>>(ws, out);
}
